// VQVAE_28003186770188
// MI455X (gfx1250) — matrix operand order repaired and checked by reading, not compiled, not run on hardware
//
#include <hip/hip_runtime.h>
#include <hip/hip_bf16.h>

// ---------------------------------------------------------------------------
// VQ-VAE forward for MI455X (gfx1250, wave32).
//   k0: emb -> bf16 copy + 0.5*||e||^2 per code; zero loss slot
//   k1: pre 1x1 conv (fp32), emit zf (fp32) + zf (bf16)
//   k2: distance matmul via v_wmma_f32_16x16x32_bf16 + fused argmin,
//       codebook tiles double-buffered in LDS via global_load_async_to_lds,
//       8 waves/block share B tiles, 2 token-tiles per wave (4 WMMA / B-frag)
//   k3: gather q = emb[idx], post 1x1 conv, loss = 1.25*mean((ze-q)^2)
// ---------------------------------------------------------------------------

typedef __attribute__((ext_vector_type(16))) __bf16 v16bf;
typedef __attribute__((ext_vector_type(8)))  __bf16 v8bf;
typedef __attribute__((ext_vector_type(8)))  float  v8f;

// async-to-LDS builtin operand types (from hipcc diagnostic: vector_size int4,
// AS1 for the global source, AS3 for the LDS destination)
typedef int v4i_ __attribute__((vector_size(16)));
typedef __attribute__((address_space(1))) v4i_* gv4i_p;
typedef __attribute__((address_space(3))) v4i_* lv4i_p;

#define C_DIM 64
#define K_DIM 8192
#define N_TOK 32768      // 8 * 64 * 64
#define HW    4096       // 64 * 64
#define TOTAL_ELEMS 2097152.0f  // 8*64*64*64

#define KT      32       // codes per LDS tile
#define ROWPAD  72       // bf16 elems per padded LDS row (144 B -> conflict-free)

#if __has_builtin(__builtin_amdgcn_global_load_async_to_lds_b128)
#define VQ_ASYNC 1
#else
#define VQ_ASYNC 0
#endif

// ---------------- k0: codebook prep ----------------------------------------
__global__ __launch_bounds__(256) void vq_prep_emb(
    const float* __restrict__ emb, __bf16* __restrict__ ebf,
    float* __restrict__ halfE, float* __restrict__ lossSlot)
{
    int r = blockIdx.x * 256 + threadIdx.x;
    if (r == 0) *lossSlot = 0.0f;
    if (r >= K_DIM) return;
    const float* er = emb + (size_t)r * C_DIM;
    __bf16* eo = ebf + (size_t)r * C_DIM;
    float s = 0.0f;
    #pragma unroll
    for (int c = 0; c < C_DIM; ++c) {
        float v = er[c];
        s = fmaf(v, v, s);
        eo[c] = (__bf16)v;
    }
    halfE[r] = 0.5f * s;
}

// ---------------- k1: pre 1x1 conv -----------------------------------------
__global__ __launch_bounds__(256) void vq_preconv(
    const float* __restrict__ z, const float* __restrict__ pre_w,
    const float* __restrict__ pre_b,
    float* __restrict__ zf, __bf16* __restrict__ zbf)
{
    __shared__ float sW[C_DIM * C_DIM];
    __shared__ float sB[C_DIM];
    for (int i = threadIdx.x; i < C_DIM * C_DIM; i += 256) sW[i] = pre_w[i];
    if (threadIdx.x < C_DIM) sB[threadIdx.x] = pre_b[threadIdx.x];
    __syncthreads();

    const int b  = blockIdx.x >> 4;
    const int hw = ((blockIdx.x & 15) << 8) + threadIdx.x;
    const float* zp = z + (size_t)b * C_DIM * HW + hw;    // coalesced per channel

    float zv[C_DIM];
    #pragma unroll
    for (int c = 0; c < C_DIM; ++c) zv[c] = zp[(size_t)c * HW];

    const int t = b * HW + hw;
    float*  zfo = zf  + (size_t)t * C_DIM;
    __bf16* zbo = zbf + (size_t)t * C_DIM;

    #pragma unroll 4
    for (int o = 0; o < C_DIM; ++o) {
        float acc = sB[o];
        const float* wr = &sW[o * C_DIM];
        #pragma unroll
        for (int c = 0; c < C_DIM; ++c) acc = fmaf(wr[c], zv[c], acc);
        zfo[o] = acc;
        zbo[o] = (__bf16)acc;
    }
}

// ---------------- k2: WMMA distances + fused argmin -------------------------
// Block = 8 waves, 32 tokens/wave (two 16-token A tiles). Codebook streamed
// in 32-code LDS tiles shared by all waves, double-buffered via async-to-LDS.
// Score = 0.5*||e||^2 - z.e ; running per-lane min, cross-lane reduce at end.
__global__ __launch_bounds__(256) void vq_argmin_wmma(
    const __bf16* __restrict__ zbf, const __bf16* __restrict__ ebf,
    const float* __restrict__ halfE, int* __restrict__ idxOut)
{
    __shared__ __bf16 sE[2][KT][ROWPAD];   // 2 x 4.5 KB padded code tiles
    __shared__ float  sHE[K_DIM];          // 32 KB: all half-norms

    const int tid  = threadIdx.x;
    const int lane = tid & 31;
    const int wave = tid >> 5;
    const int col  = lane & 15;    // A row / D code-column
    const int half = lane >> 4;

    // stage all 0.5*||e||^2 once per block
    for (int i = tid; i < K_DIM; i += 256) sHE[i] = halfE[i];

    // A fragments (ISA 7.12.2, 16-bit A 16x32) for two token tiles
    const int tokenBase = (blockIdx.x * 8 + wave) * 32;
    v16bf a[2][2];
    #pragma unroll
    for (int tt = 0; tt < 2; ++tt) {
        const __bf16* zrow = zbf + (size_t)(tokenBase + tt * 16 + col) * C_DIM;
        v8bf p0 = *(const v8bf*)(zrow +      8 * half);
        v8bf p1 = *(const v8bf*)(zrow + 16 + 8 * half);
        v8bf p2 = *(const v8bf*)(zrow + 32 + 8 * half);
        v8bf p3 = *(const v8bf*)(zrow + 48 + 8 * half);
        #pragma unroll
        for (int i = 0; i < 8; ++i) {
            a[tt][0][i] = p0[i]; a[tt][0][i + 8] = p1[i];
            a[tt][1][i] = p2[i]; a[tt][1][i + 8] = p3[i];
        }
    }

    float bestVal[2][8];
    int   bestIdx[2][8];
    #pragma unroll
    for (int tt = 0; tt < 2; ++tt)
        #pragma unroll
        for (int v = 0; v < 8; ++v) { bestVal[tt][v] = 3.4e38f; bestIdx[tt][v] = 0; }

    // per-thread staging slice: one b128 per thread per tile (256*16B = 4KB)
    const int sr = tid >> 3;          // code row 0..31
    const int sc = (tid & 7) * 8;     // element offset within row (16B chunks)
    const int ITERS = K_DIM / KT;     // 256

    auto stage = [&](int it, int bufi) {
        const __bf16* gsrc = ebf + (size_t)(it * KT + sr) * C_DIM + sc;
        __bf16* ldst = &sE[bufi][sr][sc];
#if VQ_ASYNC
        __builtin_amdgcn_global_load_async_to_lds_b128(
            (gv4i_p)(void*)gsrc, (lv4i_p)(void*)ldst, 0, 0);
#else
        v8bf v = *(const v8bf*)gsrc;
        *(v8bf*)ldst = v;
#endif
    };

    stage(0, 0);
    for (int it = 0; it < ITERS; ++it) {
        const int cb = it & 1;
        if (it + 1 < ITERS) {
            stage(it + 1, cb ^ 1);
#if VQ_ASYNC
#if __has_builtin(__builtin_amdgcn_s_wait_asynccnt)
            __builtin_amdgcn_s_wait_asynccnt(1);   // tile `it` landed in LDS
#else
            asm volatile("s_wait_asynccnt 0x1" ::: "memory");
#endif
        } else {
#if __has_builtin(__builtin_amdgcn_s_wait_asynccnt)
            __builtin_amdgcn_s_wait_asynccnt(0);
#else
            asm volatile("s_wait_asynccnt 0x0" ::: "memory");
#endif
#endif
        }
        {
            int pit = it + 2 < ITERS ? it + 2 : ITERS - 1;   // lookahead hint
            __builtin_prefetch((const void*)(ebf + (size_t)(pit * KT + sr) * C_DIM + sc), 0, 1);
        }
        __syncthreads();   // all waves' portions of tile `it` visible

        const int kbase = it * KT;
        #pragma unroll
        for (int sub = 0; sub < 2; ++sub) {
            // B fragment (32x16): lane = code column, channels in two 8-runs (table order)
            const __bf16* brow = &sE[cb][sub * 16 + col][8 * half];
            v16bf b0, b1;
            {
                v8bf q0 = *(const v8bf*)(brow);
                v8bf q1 = *(const v8bf*)(brow + 16);
                v8bf q2 = *(const v8bf*)(brow + 32);
                v8bf q3 = *(const v8bf*)(brow + 48);
                #pragma unroll
                for (int i = 0; i < 8; ++i) {
                    b0[i] = q0[i]; b0[i + 8] = q1[i];
                    b1[i] = q2[i]; b1[i + 8] = q3[i];
                }
            }
            const float he   = sHE[kbase + sub * 16 + col];
            const int   code = kbase + sub * 16 + col;
            #pragma unroll
            for (int tt = 0; tt < 2; ++tt) {
                v8f acc = {};
                acc = __builtin_amdgcn_wmma_f32_16x16x32_bf16(false, a[tt][0], false, b0,
                                                              (short)0, acc, false, false);
                acc = __builtin_amdgcn_wmma_f32_16x16x32_bf16(false, a[tt][1], false, b1,
                                                              (short)0, acc, false, false);
                #pragma unroll
                for (int v = 0; v < 8; ++v) {
                    float s = he - acc[v];
                    bool better = s < bestVal[tt][v];   // strict < keeps lowest index
                    bestVal[tt][v] = better ? s : bestVal[tt][v];
                    bestIdx[tt][v] = better ? code : bestIdx[tt][v];
                }
            }
        }
        __syncthreads();   // buffer cb free for reuse next-next iteration
    }

    // Reduce over 16 code-columns in each lane-half; token = tt*16 + v + 8*half.
    #pragma unroll
    for (int tt = 0; tt < 2; ++tt) {
        #pragma unroll
        for (int v = 0; v < 8; ++v) {
            float bv = bestVal[tt][v];
            int   bi = bestIdx[tt][v];
            #pragma unroll
            for (int m = 1; m < 16; m <<= 1) {   // xor masks stay inside 16-lane halves
                float ov = __shfl_xor(bv, m, 32);
                int   oi = __shfl_xor(bi, m, 32);
                if (ov < bv || (ov == bv && oi < bi)) { bv = ov; bi = oi; }
            }
            if (col == 0) idxOut[tokenBase + tt * 16 + v + 8 * half] = bi;
        }
    }
}

// ---------------- k3: gather + post conv + loss -----------------------------
__global__ __launch_bounds__(256) void vq_postconv(
    const int* __restrict__ idx, const float* __restrict__ emb,
    const float* __restrict__ zf, const float* __restrict__ post_w,
    const float* __restrict__ post_b,
    float* __restrict__ out, float* __restrict__ lossSlot)
{
    __shared__ float sW[C_DIM * C_DIM];
    __shared__ float sB[C_DIM];
    __shared__ float sRed[256];
    for (int i = threadIdx.x; i < C_DIM * C_DIM; i += 256) sW[i] = post_w[i];
    if (threadIdx.x < C_DIM) sB[threadIdx.x] = post_b[threadIdx.x];
    __syncthreads();

    const int b  = blockIdx.x >> 4;
    const int hw = ((blockIdx.x & 15) << 8) + threadIdx.x;
    const int t  = b * HW + hw;

    const int id = idx[t];
    const float* q  = emb + (size_t)id * C_DIM;
    const float* zr = zf  + (size_t)t  * C_DIM;

    float qv[C_DIM];
    float lacc = 0.0f;
    #pragma unroll
    for (int c = 0; c < C_DIM; ++c) {
        float qc = q[c];
        qv[c] = qc;
        float d = zr[c] - qc;
        lacc = fmaf(d, d, lacc);
    }

    float* op = out + (size_t)b * C_DIM * HW + hw;   // coalesced per channel
    #pragma unroll 4
    for (int o = 0; o < C_DIM; ++o) {
        float acc = sB[o];
        const float* wr = &sW[o * C_DIM];
        #pragma unroll
        for (int c = 0; c < C_DIM; ++c) acc = fmaf(wr[c], qv[c], acc);
        op[(size_t)o * HW] = acc;
    }

    // loss = (1 + 0.25) * mean((ze - q)^2)
    sRed[threadIdx.x] = lacc;
    __syncthreads();
    for (int s = 128; s > 0; s >>= 1) {
        if (threadIdx.x < s) sRed[threadIdx.x] += sRed[threadIdx.x + s];
        __syncthreads();
    }
    if (threadIdx.x == 0)
        atomicAdd(lossSlot, sRed[0] * (1.25f / TOTAL_ELEMS));
}

// ---------------------------------------------------------------------------
extern "C" void kernel_launch(void* const* d_in, const int* in_sizes, int n_in,
                              void* d_out, int out_size, void* d_ws, size_t ws_size,
                              hipStream_t stream) {
    const float* z      = (const float*)d_in[0];
    const float* pre_w  = (const float*)d_in[1];
    const float* pre_b  = (const float*)d_in[2];
    const float* emb    = (const float*)d_in[3];
    const float* post_w = (const float*)d_in[4];
    const float* post_b = (const float*)d_in[5];

    float* out      = (float*)d_out;
    float* lossSlot = out + (out_size - 1);   // loss is the trailing scalar

    char* ws = (char*)d_ws;
    float*  zf    = (float*) (ws + 0);            //  8,388,608 B  [N,64] fp32
    __bf16* zbf   = (__bf16*)(ws + 8388608);      //  4,194,304 B  [N,64] bf16
    __bf16* ebf   = (__bf16*)(ws + 12582912);     //  1,048,576 B  [K,64] bf16
    float*  halfE = (float*) (ws + 13631488);     //     32,768 B  [K]
    int*    idxB  = (int*)   (ws + 13664256);     //    131,072 B  [N]

    vq_prep_emb<<<K_DIM / 256, 256, 0, stream>>>(emb, ebf, halfE, lossSlot);
    vq_preconv<<<N_TOK / 256, 256, 0, stream>>>(z, pre_w, pre_b, zf, zbf);
    vq_argmin_wmma<<<N_TOK / 256, 256, 0, stream>>>(zbf, ebf, halfE, idxB);
    vq_postconv<<<N_TOK / 256, 256, 0, stream>>>(idxB, emb, zf, post_w, post_b,
                                                 out, lossSlot);
}
